// NlpGan_80960133529845
// MI455X (gfx1250) — matrix operand order repaired and checked by reading, not compiled, not run on hardware
//
#include <hip/hip_runtime.h>

// ---------------------------------------------------------------------------
// Types for CDNA5 WMMA (gfx1250, wave32)
// ---------------------------------------------------------------------------
typedef __attribute__((ext_vector_type(16))) __bf16 bf16x16;
typedef __attribute__((ext_vector_type(8)))  __bf16 bf16x8;
typedef __attribute__((ext_vector_type(8)))  float  v8f;

#define S_  256
#define B_  512
#define D_  128
#define H_  256   // HG == HD == 256
#define H4_ 1024
#define IV_ 0.1f

__device__ __forceinline__ v8f wmma_bf16(bf16x16 a, bf16x16 b, v8f c) {
  return __builtin_amdgcn_wmma_f32_16x16x32_bf16(
      /*neg_a=*/false, a, /*neg_b=*/false, b,
      /*c_mod=*/(short)0, c, /*reuse_a=*/false, /*reuse_b=*/false);
}

// Hardware tanh (V_TANH_F32 is a CDNA5 trans op); fall back to libm if the
// builtin is unavailable on this toolchain.
#if defined(__has_builtin)
#if __has_builtin(__builtin_amdgcn_tanhf)
#define HAVE_HW_TANH 1
#endif
#endif
__device__ __forceinline__ float fast_tanh(float x) {
#ifdef HAVE_HW_TANH
  return __builtin_amdgcn_tanhf(x);
#else
  return tanhf(x);
#endif
}
// sigmoid(x) = 0.5*tanh(x/2) + 0.5  (one trans op instead of exp+rcp chain)
__device__ __forceinline__ float sigm(float x) {
  return __builtin_fmaf(0.5f, fast_tanh(0.5f * x), 0.5f);
}

// A-fragment (16x32, bf16, row-major source, stride lda elements).
// ISA layout: lanes 0-15: M=lane, K runs {0..7,16..23}; lanes 16-31: M=lane-16,
// K runs {8..15,24..31}.  Two contiguous 16B loads per lane.
__device__ __forceinline__ bf16x16 load_a_bf16(const __bf16* __restrict__ base,
                                               int lda, int lane) {
  const __bf16* p = base + (lane & 15) * lda + ((lane >> 4) << 3);
  bf16x16 a;
  ((bf16x8*)&a)[0] = *(const bf16x8*)p;
  ((bf16x8*)&a)[1] = *(const bf16x8*)(p + 16);
  return a;
}

// Same A-fragment but sourced from f32 (optionally + iv*noise), cvt to bf16.
// NOISE is compile-time so there is no branch / exec manipulation in the loop.
// float2 loads only (g_out pointer is just 8-byte aligned).
template <bool NOISE>
__device__ __forceinline__ bf16x16 load_a_f32(const float* __restrict__ base,
                                              const float* __restrict__ nbase,
                                              float iv, int lda, int lane) {
  const int ro = (lane & 15) * lda + ((lane >> 4) << 3);
  bf16x16 a;
#pragma unroll
  for (int h = 0; h < 2; ++h) {
#pragma unroll
    for (int v = 0; v < 4; ++v) {
      const int o = ro + h * 16 + v * 2;
      float2 f = *(const float2*)(base + o);
      if constexpr (NOISE) {
        float2 nz = *(const float2*)(nbase + o);
        f.x += iv * nz.x;
        f.y += iv * nz.y;
      }
      a[h * 8 + v * 2]     = (__bf16)f.x;
      a[h * 8 + v * 2 + 1] = (__bf16)f.y;
    }
  }
  return a;
}

// B-fragment (32x16) from transposed bf16 weights wT[N][K] (stride ldk).
// Lane holds column N = col0 + (lane&15); K runs {0..7,16..23} (lanes 0-15) / {8..15,24..31} (lanes 16-31).
__device__ __forceinline__ bf16x16 load_b(const __bf16* __restrict__ wT,
                                          int col0, int ldk, int kc, int lane) {
  const __bf16* p = wT + (size_t)(col0 + (lane & 15)) * ldk + kc + ((lane >> 4) << 3);
  bf16x16 b;
  ((bf16x8*)&b)[0] = *(const bf16x8*)p;
  ((bf16x8*)&b)[1] = *(const bf16x8*)(p + 16);
  return b;
}

// ---------------------------------------------------------------------------
// One LSTM time step: gates = x@Wx + h@Wh + b, fused cell update.
// Grid: 128 blocks x 128 threads (4 waves). Wave -> (mtile, jtile):
//   jtile = blockIdx.x>>3 (16 hidden-col tiles), mtile = (blockIdx.x&7)*4+wave.
// Each wave: 4 gate tiles of 16x16 f32 over K = 128 (x) + 256 (h) -> 48 WMMA.
// ---------------------------------------------------------------------------
template <bool NOISE>
__global__ __launch_bounds__(128) void lstm_step(
    const float* __restrict__ x, const float* __restrict__ xn, float iv,
    const __bf16* __restrict__ hprev, __bf16* __restrict__ hout,
    const __bf16* __restrict__ wxT, const __bf16* __restrict__ whT,
    const float* __restrict__ bias, const float* __restrict__ c0,
    float* __restrict__ c, int first) {
  const int lane = threadIdx.x & 31;
  const int wave = threadIdx.x >> 5;
  const int jt   = blockIdx.x >> 3;                  // 0..15
  const int mt   = ((blockIdx.x & 7) << 2) | wave;   // 0..31
  const int row0 = mt << 4;

  const v8f z8 = {0.f, 0.f, 0.f, 0.f, 0.f, 0.f, 0.f, 0.f};
  v8f acc[4];
#pragma unroll
  for (int g = 0; g < 4; ++g) acc[g] = z8;

  // x-part: K = 128
#pragma unroll
  for (int kc = 0; kc < D_; kc += 32) {
    bf16x16 a = load_a_f32<NOISE>(x + row0 * D_ + kc, xn + row0 * D_ + kc,
                                  iv, D_, lane);
#pragma unroll
    for (int g = 0; g < 4; ++g)
      acc[g] = wmma_bf16(a, load_b(wxT, g * H_ + jt * 16, D_, kc, lane), acc[g]);
  }
  // h-part: K = 256
#pragma unroll
  for (int kc = 0; kc < H_; kc += 32) {
    bf16x16 a = load_a_bf16(hprev + row0 * H_ + kc, H_, lane);
#pragma unroll
    for (int g = 0; g < 4; ++g)
      acc[g] = wmma_bf16(a, load_b(whT, g * H_ + jt * 16, H_, kc, lane), acc[g]);
  }

  // Fused cell update (gate order i,f,g,o as columns [0:H],[H:2H],[2H:3H],[3H:4H])
  const int nloc  = lane & 15;
  const int mbase = (lane >> 4) << 3;
  const int col   = jt * 16 + nloc;
  const float bI = bias[col], bF = bias[H_ + col];
  const float bG = bias[2 * H_ + col], bO = bias[3 * H_ + col];
#pragma unroll
  for (int r = 0; r < 8; ++r) {
    const int m = row0 + mbase + r;
    const float ig = sigm(acc[0][r] + bI);
    const float fg = sigm(acc[1][r] + bF);
    const float gg = fast_tanh(acc[2][r] + bG);
    const float og = sigm(acc[3][r] + bO);
    const float cp = first ? c0[col] : c[m * H_ + col];
    const float cn = fg * cp + ig * gg;
    c[m * H_ + col] = cn;
    hout[m * H_ + col] = (__bf16)(og * fast_tanh(cn));
  }
}

// ---------------------------------------------------------------------------
// Generator readout: g = tanh(hs@Wro + bro), per-row L2 normalize, f32 out.
// One wave owns a 16x128 row block -> 64 WMMA + in-wave shfl reduction.
// Grid: 1024 blocks x 256 threads (8 waves) = 8192 row tiles of [S*B, 256].
// ---------------------------------------------------------------------------
__global__ __launch_bounds__(256) void gen_readout(
    const __bf16* __restrict__ hs, const __bf16* __restrict__ wroT,
    const float* __restrict__ bro, float* __restrict__ gout) {
  const int lane = threadIdx.x & 31;
  const int mt   = (blockIdx.x << 3) | (threadIdx.x >> 5); // 0..8191
  const long row0 = (long)mt << 4;

  const v8f z8 = {0.f, 0.f, 0.f, 0.f, 0.f, 0.f, 0.f, 0.f};
  v8f acc[8];
#pragma unroll
  for (int nt = 0; nt < 8; ++nt) acc[nt] = z8;

#pragma unroll
  for (int kc = 0; kc < H_; kc += 32) {
    bf16x16 a = load_a_bf16(hs + row0 * H_ + kc, H_, lane);
#pragma unroll
    for (int nt = 0; nt < 8; ++nt)
      acc[nt] = wmma_bf16(a, load_b(wroT, nt * 16, H_, kc, lane), acc[nt]);
  }

  const int nloc = lane & 15;
  const int mbase = (lane >> 4) << 3;
  float rs[8];
#pragma unroll
  for (int r = 0; r < 8; ++r) rs[r] = 0.f;
#pragma unroll
  for (int nt = 0; nt < 8; ++nt) {
    const float bb = bro[nt * 16 + nloc];
#pragma unroll
    for (int r = 0; r < 8; ++r) {
      const float v = fast_tanh(acc[nt][r] + bb);
      acc[nt][r] = v;
      rs[r] += v * v;
    }
  }
  // Reduce over the 16 lanes holding one row (xor masks < 16 keep halves apart)
#pragma unroll
  for (int r = 0; r < 8; ++r) {
    float s = rs[r];
    s += __shfl_xor(s, 1, 32);
    s += __shfl_xor(s, 2, 32);
    s += __shfl_xor(s, 4, 32);
    s += __shfl_xor(s, 8, 32);
    rs[r] = rsqrtf(fmaxf(s, 1e-12f));
  }
#pragma unroll
  for (int nt = 0; nt < 8; ++nt)
#pragma unroll
    for (int r = 0; r < 8; ++r)
      gout[(row0 + mbase + r) * D_ + nt * 16 + nloc] = acc[nt][r] * rs[r];
}

// ---------------------------------------------------------------------------
// Small helper kernels
// ---------------------------------------------------------------------------
__global__ void transpose_to_bf16(const float* __restrict__ in,
                                  __bf16* __restrict__ out, int K, int N) {
  const int idx = blockIdx.x * blockDim.x + threadIdx.x; // over N*K outputs [n][k]
  if (idx >= K * N) return;
  const int n = idx / K, k = idx % K;
  out[idx] = (__bf16)in[k * N + n];
}

__global__ void bcast_bf16(const float* __restrict__ v, __bf16* __restrict__ out,
                           int H, int total) {
  const int idx = blockIdx.x * blockDim.x + threadIdx.x;
  if (idx >= total) return;
  out[idx] = (__bf16)v[idx % H];
}

__global__ void disc_logits(const __bf16* __restrict__ h,
                            const float* __restrict__ w,
                            const float* __restrict__ b,
                            float* __restrict__ out) {
  const int i = blockIdx.x * blockDim.x + threadIdx.x;
  if (i >= B_) return;
  float s = 0.f;
#pragma unroll 8
  for (int k = 0; k < H_; ++k) s += (float)h[i * H_ + k] * w[k];
  out[i] = s + b[0];
}

__global__ __launch_bounds__(512) void gan_loss(const float* __restrict__ lf,
                                                const float* __restrict__ lr,
                                                float* __restrict__ out,
                                                int goutN) {
  __shared__ float sg[512], sd[512], sa[512];
  const int b = threadIdx.x;
  const float f = lf[b], r = lr[b];
  auto sp = [](float x) { return fmaxf(x, 0.f) + log1pf(__expf(-fabsf(x))); };
  sg[b] = sp(-f);
  sd[b] = sp(f) + sp(-r);
  // round(sigmoid(x)) == 0  <=>  x <= 0 (half rounds to even 0);  ==1 <=> x > 0
  sa[b] = (f <= 0.f ? 1.f : 0.f) + (r > 0.f ? 1.f : 0.f);
  __syncthreads();
  for (int s = 256; s > 0; s >>= 1) {
    if (b < s) { sg[b] += sg[b + s]; sd[b] += sd[b + s]; sa[b] += sa[b + s]; }
    __syncthreads();
  }
  if (b == 0) {
    out[0] = sd[0] / (2.f * B_);         // d_loss
    out[1] = sg[0] / (float)B_;          // g_loss
    out[2 + goutN] = sa[0] / (2.f * B_); // d_acc
  }
}

// ---------------------------------------------------------------------------
// Host orchestration
// ---------------------------------------------------------------------------
extern "C" void kernel_launch(void* const* d_in, const int* in_sizes, int n_in,
                              void* d_out, int out_size, void* d_ws, size_t ws_size,
                              hipStream_t stream) {
  const float* inputs     = (const float*)d_in[0];
  const float* z          = (const float*)d_in[1];
  const float* noise_fake = (const float*)d_in[2];
  const float* noise_real = (const float*)d_in[3];
  const float* Wx_g  = (const float*)d_in[4];
  const float* Wh_g  = (const float*)d_in[5];
  const float* b_g   = (const float*)d_in[6];
  const float* c0_g  = (const float*)d_in[7];
  const float* h0_g  = (const float*)d_in[8];
  const float* Wro_g = (const float*)d_in[9];
  const float* bro_g = (const float*)d_in[10];
  const float* Wx_d  = (const float*)d_in[11];
  const float* Wh_d  = (const float*)d_in[12];
  const float* b_d   = (const float*)d_in[13];
  const float* c0_d  = (const float*)d_in[14];
  const float* h0_d  = (const float*)d_in[15];
  const float* Wro_d = (const float*)d_in[16];
  const float* bro_d = (const float*)d_in[17];

  float* out  = (float*)d_out;
  float* gout = out + 2;                         // g_out [S,B,D] f32
  const int goutN = S_ * B_ * D_;                // 16,777,216

  // Workspace carve (256B-aligned slices)
  char* wsp = (char*)d_ws;
  size_t off = 0;
  auto carve = [&](size_t bytes) -> char* {
    char* p = wsp + off;
    off = (off + bytes + 255) & ~(size_t)255;
    return p;
  };
  const size_t BH = (size_t)B_ * H_;             // 131072
  const size_t BD = (size_t)B_ * D_;             // 65536
  __bf16* hs_g     = (__bf16*)carve((size_t)S_ * BH * 2); // 67.1 MB
  __bf16* h_init_g = (__bf16*)carve(BH * 2);
  __bf16* h_init_d = (__bf16*)carve(BH * 2);
  __bf16* hbuf0    = (__bf16*)carve(BH * 2);
  __bf16* hbuf1    = (__bf16*)carve(BH * 2);
  float*  c_g      = (float*)carve(BH * 4);
  float*  c_d      = (float*)carve(BH * 4);
  float*  logits_f = (float*)carve(B_ * 4);
  float*  logits_r = (float*)carve(B_ * 4);
  __bf16* WxT_g    = (__bf16*)carve((size_t)H4_ * D_ * 2);
  __bf16* WhT_g    = (__bf16*)carve((size_t)H4_ * H_ * 2);
  __bf16* WxT_d    = (__bf16*)carve((size_t)H4_ * D_ * 2);
  __bf16* WhT_d    = (__bf16*)carve((size_t)H4_ * H_ * 2);
  __bf16* WroT_g   = (__bf16*)carve((size_t)D_ * H_ * 2);
  (void)ws_size; (void)in_sizes; (void)n_in; (void)out_size;

  // --- Prep: transpose weights to bf16 [N][K], broadcast initial h ---
  auto tgrid = [](int n) { return dim3((n + 255) / 256); };
  transpose_to_bf16<<<tgrid(H4_ * D_), 256, 0, stream>>>(Wx_g, WxT_g, D_, H4_);
  transpose_to_bf16<<<tgrid(H4_ * H_), 256, 0, stream>>>(Wh_g, WhT_g, H_, H4_);
  transpose_to_bf16<<<tgrid(H4_ * D_), 256, 0, stream>>>(Wx_d, WxT_d, D_, H4_);
  transpose_to_bf16<<<tgrid(H4_ * H_), 256, 0, stream>>>(Wh_d, WhT_d, H_, H4_);
  transpose_to_bf16<<<tgrid(D_ * H_), 256, 0, stream>>>(Wro_g, WroT_g, H_, D_);
  bcast_bf16<<<tgrid((int)BH), 256, 0, stream>>>(h0_g, h_init_g, H_, (int)BH);
  bcast_bf16<<<tgrid((int)BH), 256, 0, stream>>>(h0_d, h_init_d, H_, (int)BH);

  // --- Generator rollout over z (no noise: template removes that path) ---
  for (int t = 0; t < S_; ++t) {
    const __bf16* hp = t ? hs_g + (size_t)(t - 1) * BH : h_init_g;
    lstm_step<false><<<dim3(128), dim3(128), 0, stream>>>(
        z + (size_t)t * BD, nullptr, 0.f, hp, hs_g + (size_t)t * BH,
        WxT_g, WhT_g, b_g, c0_g, c_g, t == 0);
  }
  // --- Readout: tanh + L2 normalize -> g_out in d_out ---
  gen_readout<<<dim3(1024), dim3(256), 0, stream>>>(hs_g, WroT_g, bro_g, gout);

  // --- Discriminator on fake (g_out + IV*noise_fake) ---
  for (int t = 0; t < S_; ++t) {
    const __bf16* hp = t ? (((t - 1) & 1) ? hbuf1 : hbuf0) : h_init_d;
    __bf16* ho = (t & 1) ? hbuf1 : hbuf0;
    lstm_step<true><<<dim3(128), dim3(128), 0, stream>>>(
        gout + (size_t)t * BD, noise_fake + (size_t)t * BD, IV_, hp, ho,
        WxT_d, WhT_d, b_d, c0_d, c_d, t == 0);
  }
  disc_logits<<<dim3(2), dim3(256), 0, stream>>>(hbuf1, Wro_d, bro_d, logits_f);

  // --- Discriminator on real (inputs + IV*noise_real) ---
  for (int t = 0; t < S_; ++t) {
    const __bf16* hp = t ? (((t - 1) & 1) ? hbuf1 : hbuf0) : h_init_d;
    __bf16* ho = (t & 1) ? hbuf1 : hbuf0;
    lstm_step<true><<<dim3(128), dim3(128), 0, stream>>>(
        inputs + (size_t)t * BD, noise_real + (size_t)t * BD, IV_, hp, ho,
        WxT_d, WhT_d, b_d, c0_d, c_d, t == 0);
  }
  disc_logits<<<dim3(2), dim3(256), 0, stream>>>(hbuf1, Wro_d, bro_d, logits_r);

  // --- Losses / accuracy ---
  gan_loss<<<dim3(1), dim3(512), 0, stream>>>(logits_f, logits_r, out, goutN);
}
